// MCSHeteroGNN_5325759447493
// MI455X (gfx1250) — matrix operand order repaired and checked by reading, not compiled, not run on hardware
//
#include <hip/hip_runtime.h>
#include <hip/hip_bf16.h>

typedef __attribute__((ext_vector_type(16))) _Float16 v16h;
typedef __attribute__((ext_vector_type(8)))  float    v8f;

#define HH 64
#define HEADS 4
#define CDIM 16

static __host__ __device__ inline int cdiv(int a, int b) { return (a + b - 1) / b; }

// ---------------------------------------------------------------------------
// Fill helpers
// ---------------------------------------------------------------------------
__global__ void k_fill_f32(float* p, float v, int n) {
    int i = blockIdx.x * blockDim.x + threadIdx.x;
    if (i < n) p[i] = v;
}
__global__ void k_fill_u32(unsigned* p, unsigned v, int n) {
    int i = blockIdx.x * blockDim.x + threadIdx.x;
    if (i < n) p[i] = v;
}

// ---------------------------------------------------------------------------
// Input linear: h = relu(x @ W + b), x:[N,F], W:[F,64]
// ---------------------------------------------------------------------------
__global__ void k_in_linear(const float* __restrict__ x, const float* __restrict__ W,
                            const float* __restrict__ b, float* __restrict__ h,
                            int N, int F) {
    int n = blockIdx.x * blockDim.x + threadIdx.x;
    if (n >= N) return;
    float xi[8];
    for (int k = 0; k < F; ++k) xi[k] = x[(size_t)n * F + k];
    for (int c = 0; c < HH; ++c) {
        float s = b[c];
        for (int k = 0; k < F; ++k) s += xi[k] * W[k * HH + c];
        h[(size_t)n * HH + c] = fmaxf(s, 0.f);
    }
}

// ---------------------------------------------------------------------------
// Pack Wsrc (f32, [lj][64][64]) into f16 WMMA B-fragment order:
//   Wp[lj][(kt*4+nt)*32 + lane][i]  (16 contiguous halves per lane)
//   where lane L holds column N = nt*16 + L%16, K = kt*32 + (L<16?0:8) + (i&7) + (i>=8?16:0)
// So the GEMM can fetch each B fragment as one contiguous 32-byte v16h.
// ---------------------------------------------------------------------------
__global__ void k_pack_w(const float* __restrict__ W, _Float16* __restrict__ Wp, int total) {
    int idx = blockIdx.x * blockDim.x + threadIdx.x;
    if (idx >= total) return;
    int i    = idx & 15;
    int lane = (idx >> 4) & 31;
    int nt   = (idx >> 9) & 3;
    int kt   = (idx >> 11) & 1;
    int lj   = idx >> 12;
    int K    = kt * 32 + ((lane < 16) ? 0 : 8) + (i & 7) + ((i >> 3) << 4);
    int col  = nt * 16 + (lane & 15);
    Wp[idx] = (_Float16)W[(size_t)lj * HH * HH + (size_t)K * HH + col];
}

// ---------------------------------------------------------------------------
// Vdst[lj][k][hd] = sum_c Wdst[lj][k][hd*16+c] * a_dst[lj][hd][c]
// (fold dst transform + attention vector: al_d = h_dst @ Vdst)
// ---------------------------------------------------------------------------
__global__ void k_vdst(const float* __restrict__ Wdst, const float* __restrict__ adst,
                       float* __restrict__ V, int total /* L5*64*4 */) {
    int idx = blockIdx.x * blockDim.x + threadIdx.x;
    if (idx >= total) return;
    int lj = idx >> 8;           // /256
    int rem = idx & 255;
    int k = rem >> 2;
    int hd = rem & 3;
    const float* wrow = Wdst + ((size_t)(lj * HH + k)) * HH + hd * CDIM;
    const float* av   = adst + ((size_t)(lj * HEADS + hd)) * CDIM;
    float s = 0.f;
#pragma unroll
    for (int c = 0; c < CDIM; ++c) s += wrow[c] * av[c];
    V[idx] = s;
}

// ---------------------------------------------------------------------------
// WMMA GEMM: out[N,64] = h[N,64] @ W[64,64]   (f16 inputs, f32 accumulate)
// block = 128 threads = 4 waves; each wave computes a 16x64 output tile.
// A fragment (16-bit A 16x32, ISA 7.12.2): lane L: M = L%16, kgrp=(L<16)?0:8,
//   halves = two contiguous 8-float runs [kbase+kgrp, +8) and [kbase+16+kgrp, +8)
//   -> 4x float4 global loads + packed cvt.
// B fragment: contiguous 32B v16h from packed LDS (2x ds_load_b128).
// C/D layout: lane L: Ncol = L%16 ; acc[r] = row (L<16?0:8)+r.
// ---------------------------------------------------------------------------
__global__ __launch_bounds__(128)
void k_gemm64(const float* __restrict__ h, const _Float16* __restrict__ Wp,
              float* __restrict__ out, int N) {
    __shared__ __align__(32) _Float16 sW[HH * HH];   // packed fragments, 8 KB
    for (int i = threadIdx.x; i < (HH * HH) / 8; i += blockDim.x)
        ((float4*)sW)[i] = ((const float4*)Wp)[i];
    __syncthreads();

    int wave = threadIdx.x >> 5;
    int lane = threadIdx.x & 31;
    int row0 = blockIdx.x * 64 + wave * 16;
    int m    = lane & 15;
    int srow = row0 + m; if (srow > N - 1) srow = N - 1;   // clamp, no divergence
    int kgrp = (lane < 16) ? 0 : 8;

    // speculative prefetch of the next block's A rows
    __builtin_prefetch(h + ((size_t)(row0 + 64)) * HH, 0, 1);

    v8f acc[4] = {v8f{}, v8f{}, v8f{}, v8f{}};
#pragma unroll
    for (int kt = 0; kt < 2; ++kt) {
        const float* ap = h + (size_t)srow * HH + kt * 32 + kgrp;
        float4 f0 = ((const float4*)ap)[0];
        float4 f1 = ((const float4*)ap)[1];
        float4 f2 = ((const float4*)(ap + 16))[0];
        float4 f3 = ((const float4*)(ap + 16))[1];
        v16h a;
        a[0]  = (_Float16)f0.x; a[1]  = (_Float16)f0.y;
        a[2]  = (_Float16)f0.z; a[3]  = (_Float16)f0.w;
        a[4]  = (_Float16)f1.x; a[5]  = (_Float16)f1.y;
        a[6]  = (_Float16)f1.z; a[7]  = (_Float16)f1.w;
        a[8]  = (_Float16)f2.x; a[9]  = (_Float16)f2.y;
        a[10] = (_Float16)f2.z; a[11] = (_Float16)f2.w;
        a[12] = (_Float16)f3.x; a[13] = (_Float16)f3.y;
        a[14] = (_Float16)f3.z; a[15] = (_Float16)f3.w;
#pragma unroll
        for (int nt = 0; nt < 4; ++nt) {
            v16h b = *(const v16h*)(sW + (((kt * 4 + nt) * 32 + lane) << 4));
            acc[nt] = __builtin_amdgcn_wmma_f32_16x16x32_f16(
                false, a, false, b, (short)0, acc[nt], false, false);
        }
    }

    int mbase = (lane < 16) ? 0 : 8;
    int ncol  = lane & 15;
#pragma unroll
    for (int nt = 0; nt < 4; ++nt)
#pragma unroll
        for (int r = 0; r < 8; ++r) {
            int orow = row0 + mbase + r;
            if (orow < N) out[(size_t)orow * HH + nt * 16 + ncol] = acc[nt][r];
        }
}

// ---------------------------------------------------------------------------
// al_s[n,hd] = sum_c hs[n, hd*16+c] * a_src[hd,c]
// ---------------------------------------------------------------------------
__global__ void k_al_s(const float* __restrict__ hs, const float* __restrict__ asrc,
                       float* __restrict__ als, int N4) {
    int idx = blockIdx.x * blockDim.x + threadIdx.x;
    if (idx >= N4) return;
    int n = idx >> 2, hd = idx & 3;
    const float* hp = hs + (size_t)n * HH + hd * CDIM;
    const float* ap = asrc + hd * CDIM;
    float s = 0.f;
#pragma unroll
    for (int c = 0; c < CDIM; ++c) s += hp[c] * ap[c];
    als[idx] = s;
}

// ---------------------------------------------------------------------------
// al_d[n,hd] = sum_k h[n,k] * V[k,hd]
// ---------------------------------------------------------------------------
__global__ void k_al_d(const float* __restrict__ h, const float* __restrict__ V,
                       float* __restrict__ ald, int N) {
    int n = blockIdx.x * blockDim.x + threadIdx.x;
    if (n >= N) return;
    float a0 = 0.f, a1 = 0.f, a2 = 0.f, a3 = 0.f;
    const float* hp = h + (size_t)n * HH;
    for (int k = 0; k < HH; ++k) {
        float hv = hp[k];
        const float* v = V + k * 4;
        a0 += hv * v[0]; a1 += hv * v[1]; a2 += hv * v[2]; a3 += hv * v[3];
    }
    ald[(size_t)n * 4 + 0] = a0; ald[(size_t)n * 4 + 1] = a1;
    ald[(size_t)n * 4 + 2] = a2; ald[(size_t)n * 4 + 3] = a3;
}

// ---------------------------------------------------------------------------
// Segment softmax, pass 1: ordered-uint atomicMax of leaky_relu logits.
// ---------------------------------------------------------------------------
__device__ inline float leaky02(float a) { return a > 0.f ? a : 0.2f * a; }

__global__ void k_edge_max(const int* __restrict__ src, const int* __restrict__ dst,
                           const float* __restrict__ als, const float* __restrict__ ald,
                           unsigned* __restrict__ amax, int E4) {
    int idx = blockIdx.x * blockDim.x + threadIdx.x;
    if (idx >= E4) return;
    int e = idx >> 2, hd = idx & 3;
    int s = src[e], d = dst[e];
    float a = leaky02(als[(size_t)s * 4 + hd] + ald[(size_t)d * 4 + hd]);
    unsigned bits = __float_as_uint(a);
    unsigned key = (bits & 0x80000000u) ? ~bits : (bits | 0x80000000u);
    atomicMax(&amax[(size_t)d * 4 + hd], key);
}

__global__ void k_decode_max(unsigned* __restrict__ amax, float* __restrict__ out, int n) {
    int i = blockIdx.x * blockDim.x + threadIdx.x;
    if (i >= n) return;
    unsigned key = amax[i];
    unsigned bits = (key & 0x80000000u) ? (key & 0x7fffffffu) : ~key;
    out[i] = __uint_as_float(bits);
}

// pass 2: denom[d,hd] += exp(alpha - amax)
__global__ void k_edge_sum(const int* __restrict__ src, const int* __restrict__ dst,
                           const float* __restrict__ als, const float* __restrict__ ald,
                           const float* __restrict__ amax, float* __restrict__ denom, int E4) {
    int idx = blockIdx.x * blockDim.x + threadIdx.x;
    if (idx >= E4) return;
    int e = idx >> 2, hd = idx & 3;
    int s = src[e], d = dst[e];
    float a = leaky02(als[(size_t)s * 4 + hd] + ald[(size_t)d * 4 + hd]);
    atomicAdd(&denom[(size_t)d * 4 + hd], __expf(a - amax[(size_t)d * 4 + hd]));
}

// pass 3: agg[d, hd*16..] += w * hs[s, hd*16..]
__global__ void k_edge_scatter(const int* __restrict__ src, const int* __restrict__ dst,
                               const float* __restrict__ als, const float* __restrict__ ald,
                               const float* __restrict__ amax, const float* __restrict__ denom,
                               const float* __restrict__ hs, float* __restrict__ agg, int E4) {
    int idx = blockIdx.x * blockDim.x + threadIdx.x;
    if (idx >= E4) return;
    int e = idx >> 2, hd = idx & 3;
    int s = src[e], d = dst[e];
    size_t dh = (size_t)d * 4 + hd;
    float a = leaky02(als[(size_t)s * 4 + hd] + ald[dh]);
    float w = __expf(a - amax[dh]) / (denom[dh] + 1e-16f);
    const float4* hp = (const float4*)(hs + (size_t)s * HH + hd * CDIM);
    float* ap = agg + (size_t)d * HH + hd * CDIM;
#pragma unroll
    for (int q = 0; q < 4; ++q) {
        float4 v = hp[q];
        atomicAdd(ap + q * 4 + 0, v.x * w);
        atomicAdd(ap + q * 4 + 1, v.y * w);
        atomicAdd(ap + q * 4 + 2, v.z * w);
        atomicAdd(ap + q * 4 + 3, v.w * w);
    }
}

// ---------------------------------------------------------------------------
// agg init = sum of conv biases of the edge types targeting this node type
// ---------------------------------------------------------------------------
__global__ void k_agg_init(float* __restrict__ agg, const float* __restrict__ bA,
                           const float* __restrict__ bB, int N64) {
    int idx = blockIdx.x * blockDim.x + threadIdx.x;
    if (idx >= N64) return;
    int c = idx & 63;
    float v = bA[c];
    if (bB) v += bB[c];
    agg[idx] = v;
}

// ---------------------------------------------------------------------------
// h_out = relu(layernorm(agg + h) * g + b) ; one wave (32 lanes) per node
// ---------------------------------------------------------------------------
__global__ __launch_bounds__(256)
void k_ln(const float* __restrict__ agg, const float* __restrict__ h,
          const float* __restrict__ g, const float* __restrict__ b,
          float* __restrict__ out, int N) {
    int wave = threadIdx.x >> 5, lane = threadIdx.x & 31;
    int n = blockIdx.x * 8 + wave;
    if (n >= N) return;
    size_t base = (size_t)n * HH;
    float x0 = agg[base + lane] + h[base + lane];
    float x1 = agg[base + lane + 32] + h[base + lane + 32];
    float s = x0 + x1, s2 = x0 * x0 + x1 * x1;
#pragma unroll
    for (int m = 16; m >= 1; m >>= 1) {
        s  += __shfl_xor(s, m, 32);
        s2 += __shfl_xor(s2, m, 32);
    }
    float mu = s * (1.f / 64.f);
    float var = s2 * (1.f / 64.f) - mu * mu;
    float r = rsqrtf(var + 1e-5f);
    float y0 = (x0 - mu) * r * g[lane] + b[lane];
    float y1 = (x1 - mu) * r * g[lane + 32] + b[lane + 32];
    out[base + lane]      = fmaxf(y0, 0.f);
    out[base + lane + 32] = fmaxf(y1, 0.f);
}

// ---------------------------------------------------------------------------
// Host orchestration
// ---------------------------------------------------------------------------
extern "C" void kernel_launch(void* const* d_in, const int* in_sizes, int n_in,
                              void* d_out, int out_size, void* d_ws, size_t ws_size,
                              hipStream_t stream) {
    const float* x_idle    = (const float*)d_in[0];
    const float* x_quasi   = (const float*)d_in[1];
    const float* x_task    = (const float*)d_in[2];
    const float* W_in_i    = (const float*)d_in[3];
    const float* b_in_i    = (const float*)d_in[4];
    const float* W_in_q    = (const float*)d_in[5];
    const float* b_in_q    = (const float*)d_in[6];
    const float* W_in_t    = (const float*)d_in[7];
    const float* b_in_t    = (const float*)d_in[8];
    const float* Wsrc      = (const float*)d_in[9];
    const float* Wdst      = (const float*)d_in[10];
    const float* a_src     = (const float*)d_in[11];
    const float* a_dst     = (const float*)d_in[12];
    const float* conv_bias = (const float*)d_in[13];
    const float* ln_g      = (const float*)d_in[14];
    const float* ln_b      = (const float*)d_in[15];
    const int*   ei[5]     = {(const int*)d_in[16], (const int*)d_in[17],
                              (const int*)d_in[18], (const int*)d_in[19],
                              (const int*)d_in[20]};

    const int Ni = in_sizes[0] / 8;
    const int Nq = in_sizes[1] / 6;
    const int Nt = in_sizes[2] / 8;
    const int L  = in_sizes[9] / (5 * HH * HH);
    int Ecnt[5];
    for (int j = 0; j < 5; ++j) Ecnt[j] = in_sizes[16 + j] / 2;
    const int Ns_[3] = {Ni, Nq, Nt};
    int Nmax = Ni > Nq ? Ni : Nq; if (Nt > Nmax) Nmax = Nt;

    // ---- workspace carve (256B aligned) ----
    char* p = (char*)d_ws;
    auto carve = [&](size_t bytes) -> char* {
        char* r = p;
        p += (bytes + 255) & ~(size_t)255;
        return r;
    };
    float* h[3];
    h[0] = (float*)carve((size_t)Ni * HH * 4);
    h[1] = (float*)carve((size_t)Nq * HH * 4);
    h[2] = (float*)carve((size_t)Nt * HH * 4);
    float* agg[3];
    agg[0] = (float*)carve((size_t)Ni * HH * 4);
    agg[1] = (float*)carve((size_t)Nq * HH * 4);
    agg[2] = (float*)carve((size_t)Nt * HH * 4);
    float* hs     = (float*)carve((size_t)Nmax * HH * 4);
    float* als    = (float*)carve((size_t)Nmax * 4 * 4);
    float* ald    = (float*)carve((size_t)Nmax * 4 * 4);
    float* amaxf  = (float*)carve((size_t)Nmax * 4 * 4);
    float* denom  = (float*)carve((size_t)Nmax * 4 * 4);
    _Float16* W16 = (_Float16*)carve((size_t)L * 5 * HH * HH * 2);
    float* Vdst   = (float*)carve((size_t)L * 5 * HH * 4 * 4);

    const int TB = 256;

    // ---- prep: weight packing / folding, input linears ----
    {
        int n = L * 5 * HH * HH;
        k_pack_w<<<cdiv(n, TB), TB, 0, stream>>>(Wsrc, W16, n);
        int nv = L * 5 * HH * 4;
        k_vdst<<<cdiv(nv, TB), TB, 0, stream>>>(Wdst, a_dst, Vdst, nv);
        k_in_linear<<<cdiv(Ni, TB), TB, 0, stream>>>(x_idle,  W_in_i, b_in_i, h[0], Ni, 8);
        k_in_linear<<<cdiv(Nq, TB), TB, 0, stream>>>(x_quasi, W_in_q, b_in_q, h[1], Nq, 6);
        k_in_linear<<<cdiv(Nt, TB), TB, 0, stream>>>(x_task,  W_in_t, b_in_t, h[2], Nt, 8);
    }

    // edge type table: (src type, dst type)
    const int esrc[5] = {0, 0, 1, 1, 2};
    const int edst[5] = {0, 1, 0, 2, 1};

    for (int l = 0; l < L; ++l) {
        // agg init with summed conv biases per dst type
        const float* bias_l = conv_bias + (size_t)l * 5 * HH;
        k_agg_init<<<cdiv(Ni * HH, TB), TB, 0, stream>>>(agg[0], bias_l + 0 * HH, bias_l + 2 * HH, Ni * HH);
        k_agg_init<<<cdiv(Nq * HH, TB), TB, 0, stream>>>(agg[1], bias_l + 1 * HH, bias_l + 4 * HH, Nq * HH);
        k_agg_init<<<cdiv(Nt * HH, TB), TB, 0, stream>>>(agg[2], bias_l + 3 * HH, (const float*)nullptr, Nt * HH);

        for (int j = 0; j < 5; ++j) {
            const int s = esrc[j], d = edst[j];
            const int Nsj = Ns_[s], Ndj = Ns_[d];
            const int Ej = Ecnt[j], E4 = Ej * 4;
            const int* src = ei[j];
            const int* dst = ei[j] + Ej;
            const _Float16* Wj = W16 + (size_t)(l * 5 + j) * HH * HH;

            // dense transforms (WMMA) + attention logits
            k_gemm64<<<cdiv(Nsj, 64), 128, 0, stream>>>(h[s], Wj, hs, Nsj);
            k_al_s<<<cdiv(Nsj * 4, TB), TB, 0, stream>>>(hs, a_src + (size_t)(l * 5 + j) * 64, als, Nsj * 4);
            k_al_d<<<cdiv(Ndj, TB), TB, 0, stream>>>(h[d], Vdst + (size_t)(l * 5 + j) * HH * 4, ald, Ndj);

            // segment softmax over dst nodes
            k_fill_u32<<<cdiv(Ndj * 4, TB), TB, 0, stream>>>((unsigned*)amaxf, 0u, Ndj * 4);
            k_fill_f32<<<cdiv(Ndj * 4, TB), TB, 0, stream>>>(denom, 0.f, Ndj * 4);
            k_edge_max<<<cdiv(E4, TB), TB, 0, stream>>>(src, dst, als, ald, (unsigned*)amaxf, E4);
            k_decode_max<<<cdiv(Ndj * 4, TB), TB, 0, stream>>>((unsigned*)amaxf, amaxf, Ndj * 4);
            k_edge_sum<<<cdiv(E4, TB), TB, 0, stream>>>(src, dst, als, ald, amaxf, denom, E4);
            k_edge_scatter<<<cdiv(E4, TB), TB, 0, stream>>>(src, dst, als, ald, amaxf, denom, hs, agg[d], E4);
        }

        // residual + layernorm + relu (last layer writes straight to d_out)
        float* fout = (float*)d_out;
        float* outp[3] = {fout, fout + (size_t)Ni * HH, fout + (size_t)(Ni + Nq) * HH};
        for (int t = 0; t < 3; ++t) {
            float* op = (l == L - 1) ? outp[t] : h[t];
            k_ln<<<cdiv(Ns_[t], 8), 256, 0, stream>>>(agg[t], h[t],
                                                      ln_g + (size_t)(l * 3 + t) * HH,
                                                      ln_b + (size_t)(l * 3 + t) * HH,
                                                      op, Ns_[t]);
        }
    }
}
